// FunctionPod_15839839387908
// MI455X (gfx1250) — matrix operand order repaired and checked by reading, not compiled, not run on hardware
//
#include <hip/hip_runtime.h>
#include <hip/hip_bf16.h>
#include <math.h>

// ---------------------------------------------------------------------------
// Problem constants (from reference)
// ---------------------------------------------------------------------------
constexpr int BB   = 2;
constexpr int UU   = 8;
constexpr int VV   = 512;
constexpr int CC   = 256;
constexpr int HH   = 8;
constexpr int DK   = 64;
constexpr int RR   = 256;
constexpr int CODE = 192;
constexpr float EPS    = 1e-6f;
constexpr float LN_EPS = 1e-5f;
constexpr float GAIN   = 2.5f;

constexpr int BU = BB * UU;          // 16

// ---------------------------------------------------------------------------
// Workspace layout (bytes). Total ~57 MB.
// ---------------------------------------------------------------------------
constexpr size_t OFF_MOD  = 0;                       // 5 * U*256 f32 (mq,mk,mv,m1,m2)
constexpr size_t OFF_WQT  = 40960;                   // 512*256 f16
constexpr size_t OFF_WKT  = OFF_WQT + 512*256*2;     // 512*256 f16
constexpr size_t OFF_WVT  = OFF_WKT + 512*256*2;     // 256*256 f16
constexpr size_t OFF_W1T  = OFF_WVT + 256*256*2;
constexpr size_t OFF_W2T  = OFF_W1T + 256*256*2;
constexpr size_t OFF_XQ   = 1u << 20;                          // [BU,512,256] f16
constexpr size_t OFF_XK   = OFF_XQ  + (size_t)BU*VV*CC*2;
constexpr size_t OFF_XV   = OFF_XK  + (size_t)BU*VV*CC*2;
constexpr size_t OFF_QB   = OFF_XV  + (size_t)BU*VV*CC*2;      // [BU,512,512] f16
constexpr size_t OFF_KB   = OFF_QB  + (size_t)BU*VV*512*2;
constexpr size_t OFF_VT   = OFF_KB  + (size_t)BU*VV*512*2;     // [BU,H,32,512] f16
constexpr size_t OFF_ATT  = OFF_VT  + (size_t)BU*HH*32*VV*2;   // [BU,512,256] f32
constexpr size_t OFF_AO   = OFF_ATT + (size_t)BU*VV*RR*4;      // attn_out f32
constexpr size_t OFF_YNM  = OFF_AO  + (size_t)BU*VV*RR*4;      // f16
constexpr size_t OFF_H1M  = OFF_YNM + (size_t)BU*VV*RR*2;      // f16

// ---------------------------------------------------------------------------
// WMMA helpers (CDNA5 wave32, V_WMMA_F32_16X16X32_F16)
// Fragment layouts follow CDNA5 ISA §7.12.2.
// ---------------------------------------------------------------------------
typedef __attribute__((ext_vector_type(16))) _Float16 v16h;
typedef __attribute__((ext_vector_type(8)))  _Float16 h8;
typedef __attribute__((ext_vector_type(8)))  float    v8f;

#define DEV static __device__ __forceinline__

DEV v8f wmma_f16(v16h a, v16h b, v8f c) {
  return __builtin_amdgcn_wmma_f32_16x16x32_f16(false, a, false, b, (short)0, c,
                                                false, false);
}

// A (16x32, row-major MxK): lane m=lane&15; halves 0..7 -> K = hi*8 + i,
// halves 8..15 -> K = 16 + hi*8 + i, where hi = lane>>4. Two 16B loads.
DEV v16h frag_a(const _Float16* tile, int ld, int lane) {
  const _Float16* p = tile + (size_t)(lane & 15) * ld + ((lane >> 4) << 3);
  h8 x0 = *(const h8*)(p);
  h8 x1 = *(const h8*)(p + 16);
  v16h r;
#pragma unroll
  for (int i = 0; i < 8; ++i) { r[i] = x0[i]; r[i + 8] = x1[i]; }
  return r;
}

// B (32x16, KxN) supplied as B^T row-major [N,K]: lane n=lane&15;
// halves 0..7 -> K = hi*8 + i, halves 8..15 -> K = 16 + hi*8 + i (2x16B loads).
DEV v16h frag_bT(const _Float16* tileT, int ld, int lane) {
  const _Float16* p = tileT + (size_t)(lane & 15) * ld + ((lane >> 4) << 3);
  h8 x0 = *(const h8*)(p);
  h8 x1 = *(const h8*)(p + 16);
  v16h r;
#pragma unroll
  for (int i = 0; i < 8; ++i) { r[i] = x0[i]; r[i + 8] = x1[i]; }
  return r;
}

// Batched mainloop body: rows m0..m0+15 x cols nbw..nbw+63 over K=256.
// Loads A + all 4 B fragments first (one clause, partial loadcnt waits),
// then issues the 4 independent WMMAs.
DEV void gemm16x64_k256(const _Float16* __restrict__ A, int m0,
                        const _Float16* __restrict__ Bt, int nbw, int lane,
                        v8f acc[4]) {
#pragma unroll
  for (int k0 = 0; k0 < 256; k0 += 32) {
    v16h a = frag_a(A + (size_t)m0 * 256 + k0, 256, lane);
    v16h b[4];
#pragma unroll
    for (int t = 0; t < 4; ++t)
      b[t] = frag_bT(Bt + (size_t)(nbw + t * 16) * 256 + k0, 256, lane);
#pragma unroll
    for (int t = 0; t < 4; ++t) acc[t] = wmma_f16(a, b[t], acc[t]);
  }
}

DEV float gelu_exact(float x) {
  return 0.5f * x * (1.0f + erff(x * 0.70710678118654752f));
}

// ---------------------------------------------------------------------------
// Kernel: per-u modulation vectors  m = 1 + codes[u] @ Wc   (5 of them)
// ---------------------------------------------------------------------------
__global__ __launch_bounds__(256) void k_modvec(
    const float* __restrict__ codes, const float* __restrict__ Wcq,
    const float* __restrict__ Wck, const float* __restrict__ Wcv,
    const float* __restrict__ Wc1, const float* __restrict__ Wc2,
    float* __restrict__ mod) {
  int u = blockIdx.x / 5, which = blockIdx.x % 5;
  const float* Wc = which == 0 ? Wcq : which == 1 ? Wck : which == 2 ? Wcv
                   : which == 3 ? Wc1 : Wc2;
  int c = threadIdx.x;
  float acc = 1.0f;
  const float* cr = codes + u * CODE;
#pragma unroll 4
  for (int j = 0; j < CODE; ++j) acc += cr[j] * Wc[j * 256 + c];
  mod[(size_t)which * UU * 256 + u * 256 + c] = acc;
}

// Kernel: transpose + f32->f16:  dst[n*K + k] = (f16)src[k*N + n]
__global__ __launch_bounds__(256) void k_trans(const float* __restrict__ src,
                                               _Float16* __restrict__ dst,
                                               int K, int N) {
  int i = blockIdx.x * 256 + threadIdx.x;
  if (i >= K * N) return;
  int k = i / N, n = i - k * N;
  dst[(size_t)n * K + k] = (_Float16)src[i];
}

// ---------------------------------------------------------------------------
// Kernel: LN1 + three modulations -> xq, xk, xv (f16). One row per block.
// ---------------------------------------------------------------------------
__global__ __launch_bounds__(256) void k_ln1(
    const float* __restrict__ x, const float* __restrict__ g,
    const float* __restrict__ b, const float* __restrict__ mod,
    _Float16* __restrict__ xq, _Float16* __restrict__ xk,
    _Float16* __restrict__ xv) {
  __shared__ float red[2][8];
  int row = blockIdx.x, c = threadIdx.x;
  int u = (row >> 9) & 7;
  int lane = c & 31, wid = c >> 5;
  size_t o = (size_t)row * 256 + c;
  float v = x[o];
  float s = v, sq = v * v;
#pragma unroll
  for (int m = 16; m > 0; m >>= 1) {
    s  += __shfl_xor(s, m, 32);
    sq += __shfl_xor(sq, m, 32);
  }
  if (lane == 0) { red[0][wid] = s; red[1][wid] = sq; }
  __syncthreads();
  float ts = 0.f, tq = 0.f;
#pragma unroll
  for (int i = 0; i < 8; ++i) { ts += red[0][i]; tq += red[1][i]; }
  float mean = ts * (1.f / 256.f);
  float var  = tq * (1.f / 256.f) - mean * mean;
  float xn = (v - mean) * rsqrtf(var + LN_EPS) * g[c] + b[c];
  const float* mq = mod + 0 * UU * 256 + u * 256;
  const float* mk = mod + 1 * UU * 256 + u * 256;
  const float* mv = mod + 2 * UU * 256 + u * 256;
  xq[o] = (_Float16)(xn * mq[c]);
  xk[o] = (_Float16)(xn * mk[c]);
  xv[o] = (_Float16)(xn * mv[c]);
}

// ---------------------------------------------------------------------------
// Kernel: QKV GEMMs. grid = (10 slots, 8 row-blocks, 16 bu).
// slots 0..3 -> Q (N=512), 4..7 -> K (N=512), 8..9 -> V (N=256, bias, stores
// head-transposed vT[bu][h][32][512]).
// Per block: 64 rows x 128 cols; each wave: 16 rows x 64 cols (4 accums).
// ---------------------------------------------------------------------------
__global__ __launch_bounds__(256) void k_qkv(
    const _Float16* __restrict__ xq, const _Float16* __restrict__ xk,
    const _Float16* __restrict__ xv, const _Float16* __restrict__ WqT,
    const _Float16* __restrict__ WkT, const _Float16* __restrict__ WvT,
    const float* __restrict__ bv, _Float16* __restrict__ qb,
    _Float16* __restrict__ kb, _Float16* __restrict__ vT) {
  int slot = blockIdx.x, bu = blockIdx.z;
  int lane = threadIdx.x & 31, wid = threadIdx.x >> 5;
  int m0 = blockIdx.y * 64 + (wid >> 1) * 16;
  const _Float16 *A, *Bt;
  int nbw;
  if (slot < 4)      { A = xq + (size_t)bu * VV * CC; Bt = WqT; nbw = slot * 128; }
  else if (slot < 8) { A = xk + (size_t)bu * VV * CC; Bt = WkT; nbw = (slot - 4) * 128; }
  else               { A = xv + (size_t)bu * VV * CC; Bt = WvT; nbw = (slot - 8) * 128; }
  nbw += (wid & 1) * 64;

  v8f acc[4] = {v8f{}, v8f{}, v8f{}, v8f{}};
  gemm16x64_k256(A, m0, Bt, nbw, lane, acc);

  int hi = lane >> 4, nn = lane & 15;
  if (slot < 8) {
    _Float16* ob = (slot < 4 ? qb : kb) + (size_t)bu * VV * 512;
#pragma unroll
    for (int t = 0; t < 4; ++t)
#pragma unroll
      for (int j = 0; j < 8; ++j)
        ob[(size_t)(m0 + j + 8 * hi) * 512 + nbw + t * 16 + nn] =
            (_Float16)acc[t][j];
  } else {
#pragma unroll
    for (int t = 0; t < 4; ++t)
#pragma unroll
      for (int j = 0; j < 8; ++j) {
        int n = nbw + t * 16 + nn;
        int m = m0 + j + 8 * hi;
        float val = acc[t][j] + bv[n];
        vT[(((size_t)bu * HH + (n >> 5)) * 32 + (n & 31)) * 512 + m] =
            (_Float16)val;
      }
  }
}

// ---------------------------------------------------------------------------
// Kernel: attention. grid = (8 row-blocks of 64, H, BU). 256 threads.
// Dynamic LDS: scores[64][520 f32 view / 1040 f16 view] (2080B rows) +
// aff[512] + rowscale[64].
// ---------------------------------------------------------------------------
constexpr int SROW_B = 2080;                          // padded row stride, bytes
constexpr size_t ATTN_LDS = 64 * SROW_B + 512 * 4 + 64 * 4;

__global__ __launch_bounds__(256) void k_attn(
    const _Float16* __restrict__ qb, const _Float16* __restrict__ kb,
    const _Float16* __restrict__ vT, const float* __restrict__ aff,
    float* __restrict__ att) {
  extern __shared__ __align__(16) char smem[];
  float* affL     = (float*)(smem + 64 * SROW_B);
  float* rowscale = (float*)(smem + 64 * SROW_B + 512 * 4);

  int r0 = blockIdx.x * 64, h = blockIdx.y, bu = blockIdx.z;
  int lane = threadIdx.x & 31, wid = threadIdx.x >> 5;
  int hi = lane >> 4, nn = lane & 15;

  const float* affr = aff + (size_t)bu * VV;
  for (int s = threadIdx.x; s < 512; s += 256) affL[s] = affr[s];
  __syncthreads();

  // ---- phase 1: scores = Q K^T -> LDS (f32) -------------------------------
  const _Float16* Q = qb + (size_t)bu * VV * 512 + h * 64;
  const _Float16* K = kb + (size_t)bu * VV * 512 + h * 64;
  {
    int m0 = (wid >> 1) * 16;                 // local row tile
    int ch = wid & 1;                         // column half (256 cols)
    v16h a0 = frag_a(Q + (size_t)(r0 + m0) * 512 + 0, 512, lane);
    v16h a1 = frag_a(Q + (size_t)(r0 + m0) * 512 + 32, 512, lane);
#pragma unroll 4
    for (int ct = 0; ct < 16; ++ct) {
      int n0 = ch * 256 + ct * 16;
      v16h b0 = frag_bT(K + (size_t)n0 * 512 + 0, 512, lane);
      v16h b1 = frag_bT(K + (size_t)n0 * 512 + 32, 512, lane);
      v8f c = {};
      c = wmma_f16(a0, b0, c);
      c = wmma_f16(a1, b1, c);
      float* srow = (float*)smem;
#pragma unroll
      for (int j = 0; j < 8; ++j)
        srow[(size_t)(m0 + j + 8 * hi) * 520 + n0 + nn] = c[j];
    }
  }
  __syncthreads();

  // ---- phase 2: softmax + affinity renorm, write f16 probs in place ------
  {
#pragma unroll
    for (int rr = 0; rr < 8; ++rr) {
      int rl = wid * 8 + rr;
      char* rowp = smem + (size_t)rl * SROW_B;
      int sb = lane * 16;
      float vals[16];
#pragma unroll
      for (int i = 0; i < 16; ++i) vals[i] = *(float*)(rowp + (sb + i) * 4);
      float mx = vals[0];
#pragma unroll
      for (int i = 1; i < 16; ++i) mx = fmaxf(mx, vals[i]);
#pragma unroll
      for (int m = 16; m > 0; m >>= 1) mx = fmaxf(mx, __shfl_xor(mx, m, 32));
      float s1 = 0.f, s2 = 0.f, w[16];
#pragma unroll
      for (int i = 0; i < 16; ++i) {
        float e = __expf(vals[i] - mx);
        s1 += e;
        float ww = affL[sb + i] * e;
        s2 += ww;
        w[i] = ww;
      }
#pragma unroll
      for (int m = 16; m > 0; m >>= 1) {
        s1 += __shfl_xor(s1, m, 32);
        s2 += __shfl_xor(s2, m, 32);
      }
      float ga = affL[r0 + rl];
      float f = ga / (ga * s2 + EPS * s1);   // kmw_norm = f * aff_s * e
      _Float16* hp = (_Float16*)rowp + sb;
#pragma unroll
      for (int i = 0; i < 16; ++i) hp[i] = (_Float16)w[i];
      if (lane == 0) rowscale[rl] = f;
    }
  }
  __syncthreads();

  // ---- phase 3: att = probs @ V_h  (64x512 @ 512x32) ----------------------
  // Two independent accumulator chains (even/odd k-steps) to double WMMA ILP.
  {
    int m0 = (wid >> 1) * 16;
    int n0 = (wid & 1) * 16;
    const _Float16* Vh = vT + ((size_t)bu * HH + h) * 32 * 512;
    const _Float16* P = (const _Float16*)smem;     // ld = 1040 halves
    v8f c0 = {}, c1 = {};
#pragma unroll 4
    for (int ks = 0; ks < 16; ks += 2) {
      int ka = ks * 32, kb2 = ka + 32;
      v16h a0 = frag_a(P + (size_t)m0 * 1040 + ka, 1040, lane);
      v16h b0 = frag_bT(Vh + (size_t)n0 * 512 + ka, 512, lane);
      v16h a1 = frag_a(P + (size_t)m0 * 1040 + kb2, 1040, lane);
      v16h b1 = frag_bT(Vh + (size_t)n0 * 512 + kb2, 512, lane);
      c0 = wmma_f16(a0, b0, c0);
      c1 = wmma_f16(a1, b1, c1);
    }
#pragma unroll
    for (int j = 0; j < 8; ++j) {
      int ml = m0 + j + 8 * hi;
      float val = (c0[j] + c1[j]) * rowscale[ml];
      att[((size_t)bu * VV + r0 + ml) * RR + h * 32 + n0 + nn] = val;
    }
  }
}

// ---------------------------------------------------------------------------
// Kernel: gating residual + LN2 + m1 modulation
// ---------------------------------------------------------------------------
__global__ __launch_bounds__(256) void k_ln2gate(
    const float* __restrict__ x, const float* __restrict__ att,
    const float* __restrict__ aff, const float* __restrict__ g2,
    const float* __restrict__ b2, const float* __restrict__ mod,
    float* __restrict__ attn_out, _Float16* __restrict__ ynm) {
  __shared__ float red[2][8];
  int row = blockIdx.x, c = threadIdx.x;
  int u = (row >> 9) & 7;
  int lane = c & 31, wid = c >> 5;
  size_t o = (size_t)row * 256 + c;
  float g = aff[row];
  float a = (1.0f - g) * x[o] + g * GAIN * att[o];
  attn_out[o] = a;
  float s = a, sq = a * a;
#pragma unroll
  for (int m = 16; m > 0; m >>= 1) {
    s  += __shfl_xor(s, m, 32);
    sq += __shfl_xor(sq, m, 32);
  }
  if (lane == 0) { red[0][wid] = s; red[1][wid] = sq; }
  __syncthreads();
  float ts = 0.f, tq = 0.f;
#pragma unroll
  for (int i = 0; i < 8; ++i) { ts += red[0][i]; tq += red[1][i]; }
  float mean = ts * (1.f / 256.f);
  float var  = tq * (1.f / 256.f) - mean * mean;
  float yn = (a - mean) * rsqrtf(var + LN_EPS) * g2[c] + b2[c];
  const float* m1 = mod + 3 * UU * 256 + u * 256;
  ynm[o] = (_Float16)(yn * m1[c]);
}

// ---------------------------------------------------------------------------
// Kernel: MLP layer 1:  h1m = gelu(ynm @ W1 + b1m) * m2   (f16)
// grid = (2, 8, 16)
// ---------------------------------------------------------------------------
__global__ __launch_bounds__(256) void k_mlp1(
    const _Float16* __restrict__ ynm, const _Float16* __restrict__ W1T,
    const float* __restrict__ b1m, const float* __restrict__ mod,
    _Float16* __restrict__ h1m) {
  int bu = blockIdx.z, u = bu & 7;
  int lane = threadIdx.x & 31, wid = threadIdx.x >> 5;
  int m0 = blockIdx.y * 64 + (wid >> 1) * 16;
  int nbw = blockIdx.x * 128 + (wid & 1) * 64;
  const _Float16* A = ynm + (size_t)bu * VV * RR;
  v8f acc[4] = {v8f{}, v8f{}, v8f{}, v8f{}};
  gemm16x64_k256(A, m0, W1T, nbw, lane, acc);
  int hi = lane >> 4, nn = lane & 15;
  const float* m2 = mod + 4 * UU * 256 + u * 256;
  _Float16* ob = h1m + (size_t)bu * VV * RR;
#pragma unroll
  for (int t = 0; t < 4; ++t)
#pragma unroll
    for (int j = 0; j < 8; ++j) {
      int n = nbw + t * 16 + nn;
      float hv = gelu_exact(acc[t][j] + b1m[n]);
      ob[(size_t)(m0 + j + 8 * hi) * 256 + n] = (_Float16)(hv * m2[n]);
    }
}

// ---------------------------------------------------------------------------
// Kernel: MLP layer 2 + final gated residual -> d_out (f32)
// ---------------------------------------------------------------------------
__global__ __launch_bounds__(256) void k_mlp2(
    const _Float16* __restrict__ h1m, const _Float16* __restrict__ W2T,
    const float* __restrict__ b2m, const float* __restrict__ aff,
    const float* __restrict__ attn_out, float* __restrict__ out) {
  int bu = blockIdx.z;
  int lane = threadIdx.x & 31, wid = threadIdx.x >> 5;
  int m0 = blockIdx.y * 64 + (wid >> 1) * 16;
  int nbw = blockIdx.x * 128 + (wid & 1) * 64;
  const _Float16* A = h1m + (size_t)bu * VV * RR;
  v8f acc[4] = {v8f{}, v8f{}, v8f{}, v8f{}};
  gemm16x64_k256(A, m0, W2T, nbw, lane, acc);
  int hi = lane >> 4, nn = lane & 15;
#pragma unroll
  for (int t = 0; t < 4; ++t)
#pragma unroll
    for (int j = 0; j < 8; ++j) {
      int n = nbw + t * 16 + nn;
      int m = m0 + j + 8 * hi;
      float h2 = gelu_exact(acc[t][j] + b2m[n]);
      float g = aff[(size_t)bu * VV + m];
      size_t o = ((size_t)bu * VV + m) * RR + n;
      out[o] = (1.0f - g) * attn_out[o] + GAIN * g * g * h2;
    }
}

// ---------------------------------------------------------------------------
// Host launcher
// ---------------------------------------------------------------------------
extern "C" void kernel_launch(void* const* d_in, const int* in_sizes, int n_in,
                              void* d_out, int out_size, void* d_ws,
                              size_t ws_size, hipStream_t stream) {
  (void)in_sizes; (void)n_in; (void)out_size; (void)ws_size;
  const float* variables = (const float*)d_in[0];
  const float* codes     = (const float*)d_in[1];
  const float* aff       = (const float*)d_in[2];
  const float* ln1_g     = (const float*)d_in[3];
  const float* ln1_b     = (const float*)d_in[4];
  const float* Wq        = (const float*)d_in[5];
  const float* Wcq       = (const float*)d_in[6];
  const float* Wk        = (const float*)d_in[7];
  const float* Wck       = (const float*)d_in[8];
  const float* Wv        = (const float*)d_in[9];
  const float* Wcv       = (const float*)d_in[10];
  const float* bv        = (const float*)d_in[11];
  const float* ln2_g     = (const float*)d_in[12];
  const float* ln2_b     = (const float*)d_in[13];
  const float* W1        = (const float*)d_in[14];
  const float* Wc1       = (const float*)d_in[15];
  const float* b1m       = (const float*)d_in[16];
  const float* W2        = (const float*)d_in[17];
  const float* Wc2       = (const float*)d_in[18];
  const float* b2m       = (const float*)d_in[19];
  float* out = (float*)d_out;

  char* ws = (char*)d_ws;
  float*    mod  = (float*)(ws + OFF_MOD);
  _Float16* WqT  = (_Float16*)(ws + OFF_WQT);
  _Float16* WkT  = (_Float16*)(ws + OFF_WKT);
  _Float16* WvT  = (_Float16*)(ws + OFF_WVT);
  _Float16* W1T  = (_Float16*)(ws + OFF_W1T);
  _Float16* W2T  = (_Float16*)(ws + OFF_W2T);
  _Float16* xq   = (_Float16*)(ws + OFF_XQ);
  _Float16* xk   = (_Float16*)(ws + OFF_XK);
  _Float16* xv   = (_Float16*)(ws + OFF_XV);
  _Float16* qb   = (_Float16*)(ws + OFF_QB);
  _Float16* kb   = (_Float16*)(ws + OFF_KB);
  _Float16* vT   = (_Float16*)(ws + OFF_VT);
  float*    att  = (float*)(ws + OFF_ATT);
  float*    ao   = (float*)(ws + OFF_AO);
  _Float16* ynm  = (_Float16*)(ws + OFF_YNM);
  _Float16* h1m  = (_Float16*)(ws + OFF_H1M);

  // prep: modulation vectors + f16 transposed weights
  k_modvec<<<UU * 5, 256, 0, stream>>>(codes, Wcq, Wck, Wcv, Wc1, Wc2, mod);
  k_trans<<<(256 * 512 + 255) / 256, 256, 0, stream>>>(Wq, WqT, 256, 512);
  k_trans<<<(256 * 512 + 255) / 256, 256, 0, stream>>>(Wk, WkT, 256, 512);
  k_trans<<<(256 * 256 + 255) / 256, 256, 0, stream>>>(Wv, WvT, 256, 256);
  k_trans<<<(256 * 256 + 255) / 256, 256, 0, stream>>>(W1, W1T, 256, 256);
  k_trans<<<(256 * 256 + 255) / 256, 256, 0, stream>>>(W2, W2T, 256, 256);

  // LN1 + modulation
  k_ln1<<<BU * VV, 256, 0, stream>>>(variables, ln1_g, ln1_b, mod, xq, xk, xv);

  // QKV GEMMs
  k_qkv<<<dim3(10, 8, BU), 256, 0, stream>>>(xq, xk, xv, WqT, WkT, WvT, bv,
                                             qb, kb, vT);

  // attention (dynamic LDS: 64-row score panel)
  k_attn<<<dim3(8, HH, BU), 256, ATTN_LDS, stream>>>(qb, kb, vT, aff, att);

  // gate + LN2 + m1 modulation
  k_ln2gate<<<BU * VV, 256, 0, stream>>>(variables, att, aff, ln2_g, ln2_b,
                                         mod, ao, ynm);

  // MLP
  k_mlp1<<<dim3(2, 8, BU), 256, 0, stream>>>(ynm, W1T, b1m, mod, h1m);
  k_mlp2<<<dim3(2, 8, BU), 256, 0, stream>>>(h1m, W2T, b2m, aff, ao, out);
}
